// GptOssTopKRouter_63307817943052
// MI455X (gfx1250) — matrix operand order repaired and checked by reading, not compiled, not run on hardware
//
#include <hip/hip_runtime.h>

// ---------------------------------------------------------------------------
// GPT-OSS top-k router, fused GEMM(+bias) -> top-4 -> softmax -> dense scatter
// MI455X / gfx1250: wave32, v_wmma_f32_16x16x32_bf16 with bf16 hi/lo split
// (A*B ~= Ahi*Bhi + Alo*Bhi + Ahi*Blo, f32 accum) for ~fp32 accuracy.
//
// Roofline: ~100MB mandatory HBM traffic @ 23.3TB/s -> ~4.3us floor.
// 512 blocks x 8 waves (4096 waves); one 16-expert N-tile per wave.
// A chunk converted once per block into LDS by a rotating filler wave with a
// two-stage (load || convert+store) pipeline; B fragments (pre-split bf16 in
// d_ws, L2-resident) double-buffered in registers. The hot-loop barrier is a
// raw s_wait_dscnt + s_barrier so global loads stay in flight across it.
// ---------------------------------------------------------------------------

typedef __attribute__((ext_vector_type(16))) __bf16 v16bf;
typedef __attribute__((ext_vector_type(8)))  float  v8f;

#define T_TOKENS 8192
#define H_DIM    2880
#define E_EXP    128
#define TOPK     4
#define WAVES_PER_BLOCK  8
#define BLOCK_THREADS    (WAVES_PER_BLOCK * 32)
#define TOKENS_PER_BLOCK 16
#define NUM_CHUNKS       (H_DIM / 32)          // 90 (even)
#define NEG_BIG  (-3.402823466e38f)

// LDS-visibility-only barrier: do NOT flush loadcnt (keeps global loads in
// flight across the sync, unlike __syncthreads()).
#define WG_BARRIER() \
    asm volatile("s_wait_dscnt 0x0\n\ts_barrier_signal -1\n\ts_barrier_wait -1" ::: "memory")

// ---- one-time weight split: fp32 -> bf16 hi + bf16 lo (residual) ----------
__global__ void gptoss_weight_split_kernel(const float* __restrict__ w,
                                           __bf16* __restrict__ w_hi,
                                           __bf16* __restrict__ w_lo,
                                           int n)
{
    int i = blockIdx.x * blockDim.x + threadIdx.x;
    if (i < n) {
        float f  = w[i];
        __bf16 h = (__bf16)f;
        w_hi[i]  = h;
        w_lo[i]  = (__bf16)(f - (float)h);
    }
}

__device__ __forceinline__ void split16(const float* __restrict__ f,
                                        v16bf& hi, v16bf& lo)
{
#pragma unroll
    for (int i = 0; i < 16; ++i) {
        __bf16 h = (__bf16)f[i];
        hi[i] = h;
        lo[i] = (__bf16)(f[i] - (float)h);
    }
}

// ---- fused router kernel --------------------------------------------------
__global__ __launch_bounds__(BLOCK_THREADS)
void gptoss_topk_router_kernel(const float*  __restrict__ hidden,
                               const __bf16* __restrict__ w_hi,
                               const __bf16* __restrict__ w_lo,
                               const float*  __restrict__ bias,
                               float*        __restrict__ out)
{
    // A chunk staged as WMMA-A lane fragments: [buf][hi/lo][lane][16 bf16]
    __shared__ __align__(32) __bf16 ldsA[2][2][32][16];     // 4 KB
    __shared__ float lds_logits[TOKENS_PER_BLOCK][E_EXP];   // 8 KB

    const int lane = threadIdx.x & 31;
    const int wave = threadIdx.x >> 5;     // == N-tile index (experts 16w..)
    const int half = lane >> 4;            // 0: lanes 0-15, 1: lanes 16-31
    const int l16  = lane & 15;
    const int wv   = __builtin_amdgcn_readfirstlane(wave);  // scalar branch

    const int row0 = blockIdx.x * TOKENS_PER_BLOCK;

    const float*  aRow = hidden + (size_t)(row0 + l16) * H_DIM;
    // B fragment (contiguous k order, same as A): lane group = N, lanes 0-15 K=0..15, 16-31 K=16..31
    const __bf16* bhiP = w_hi + (size_t)(wave * 16 + l16) * H_DIM + half * 16;
    const __bf16* bloP = w_lo + (size_t)(wave * 16 + l16) * H_DIM + half * 16;

    // filler pipeline registers (raw fp32 A held for one iteration)
    float4 r0, r1, r2, r3;

    // phase L: issue the global loads for one 16x32 A chunk (contiguous k order, same as B:
    // lanes 0-15: M=l16, K=0..15; lanes 16-31: M=l16, K=16..31)
    auto loadAregs = [&](int h0) {
        const float* p = aRow + h0 + half * 16;
        r0 = *(const float4*)(p + 0);
        r1 = *(const float4*)(p + 4);
        r2 = *(const float4*)(p + 8);
        r3 = *(const float4*)(p + 12);
    };
    // phase S: bf16 hi/lo split + store to LDS fragment layout
    auto storeA = [&](int buf) {
        float af[16] = { r0.x,r0.y,r0.z,r0.w, r1.x,r1.y,r1.z,r1.w,
                         r2.x,r2.y,r2.z,r2.w, r3.x,r3.y,r3.z,r3.w };
        v16bf hi, lo;
        split16(af, hi, lo);
        *(v16bf*)&ldsA[buf][0][lane][0] = hi;
        *(v16bf*)&ldsA[buf][1][lane][0] = lo;
    };

    // ---- prologue: buf0 <- chunk 0; wave1 pre-issues chunk 1's loads ------
    if (wv == 0) { loadAregs(0); storeA(0); }
    if (wv == 1) { loadAregs(32); }

    v16bf b0h = *(const v16bf*)(bhiP + 0);
    v16bf b0l = *(const v16bf*)(bloP + 0);
    v16bf b1h = *(const v16bf*)(bhiP + 32);
    v16bf b1l = *(const v16bf*)(bloP + 32);
    __builtin_prefetch(bhiP + 64, 0, 1);        // global_prefetch next B
    WG_BARRIER();

    // three independent accumulators -> no WMMA->WMMA RAW inside a chunk
    v8f acc0 = {}, acc1 = {}, acc2 = {};

    auto step = [&](int ci, int buf, v16bf& bh, v16bf& bl) {
        const int h0 = ci * 32;
        v16bf a_hi = *(const v16bf*)&ldsA[buf][0][lane][0];
        v16bf a_lo = *(const v16bf*)&ldsA[buf][1][lane][0];

        // filler store phase: publish chunk ci+1 (loaded one step ago)
        if (ci + 1 < NUM_CHUNKS && wv == ((ci + 1) & 7)) storeA(buf ^ 1);
        // filler load phase: issue chunk ci+2's global loads
        if (ci + 2 < NUM_CHUNKS && wv == ((ci + 2) & 7)) loadAregs(h0 + 64);

        acc0 = __builtin_amdgcn_wmma_f32_16x16x32_bf16(
            false, a_hi, false, bh, (short)0, acc0, false, false);
        acc1 = __builtin_amdgcn_wmma_f32_16x16x32_bf16(
            false, a_lo, false, bh, (short)0, acc1, false, false);
        acc2 = __builtin_amdgcn_wmma_f32_16x16x32_bf16(
            false, a_hi, false, bl, (short)0, acc2, false, false);

        // refill this B register set for chunk ci+2 (dummy reload at tail);
        // these loads remain in flight across the barrier below
        const int hn = (h0 + 64 < H_DIM) ? (h0 + 64) : h0;
        bh = *(const v16bf*)(bhiP + hn);
        bl = *(const v16bf*)(bloP + hn);

        WG_BARRIER();
    };

    for (int ci = 0; ci < NUM_CHUNKS; ci += 2) {
        step(ci,     0, b0h, b0l);
        step(ci + 1, 1, b1h, b1l);
    }

    // ---- epilogue: combine accumulators, bias add, spill to shared logits
    // C layout: VGPR i -> (M = i + half*8, N = l16)
    {
        int e = wave * 16 + l16;
        float bv = bias[e];
#pragma unroll
        for (int i = 0; i < 8; ++i)
            lds_logits[i + half * 8][e] = acc0[i] + acc1[i] + acc2[i] + bv;
    }
    __syncthreads();

    // ---- per-token top-4 -> softmax -> dense scatter (2 tokens per wave) --
#pragma unroll
    for (int ti = 0; ti < 2; ++ti) {
        const int t = wave * 2 + ti;
        const float* rp = &lds_logits[t][0];
        float4 v = *(const float4*)(rp + lane * 4);
        float v0 = v.x, v1 = v.y, v2 = v.z, v3 = v.w;

        float topv[TOPK];
        int   topi[TOPK];
#pragma unroll
        for (int k = 0; k < TOPK; ++k) {
            // local argmax of 4 (strict > keeps lowest index on ties)
            float mv = v0; int mi = lane * 4;
            if (v1 > mv) { mv = v1; mi = lane * 4 + 1; }
            if (v2 > mv) { mv = v2; mi = lane * 4 + 2; }
            if (v3 > mv) { mv = v3; mi = lane * 4 + 3; }
            // wave32 reduction; tie -> smaller index (matches jax top_k)
#pragma unroll
            for (int off = 16; off > 0; off >>= 1) {
                float ov = __shfl_xor(mv, off, 32);
                int   oi = __shfl_xor(mi, off, 32);
                if (ov > mv || (ov == mv && oi < mi)) { mv = ov; mi = oi; }
            }
            topv[k] = mv;
            topi[k] = mi;
            // mask out the winner in its owner lane
            int sel = mi - lane * 4;
            v0 = (sel == 0) ? NEG_BIG : v0;
            v1 = (sel == 1) ? NEG_BIG : v1;
            v2 = (sel == 2) ? NEG_BIG : v2;
            v3 = (sel == 3) ? NEG_BIG : v3;
        }

        // softmax over the 4 selected (topv[0] is the max)
        float w0 = 1.0f;
        float w1 = __expf(topv[1] - topv[0]);
        float w2 = __expf(topv[2] - topv[0]);
        float w3 = __expf(topv[3] - topv[0]);
        float inv = 1.0f / (w0 + w1 + w2 + w3);
        float ws[TOPK] = { w0 * inv, w1 * inv, w2 * inv, w3 * inv };

        // dense scatter: each lane owns experts 4*lane..4*lane+3
        float o0 = 0.f, o1 = 0.f, o2 = 0.f, o3 = 0.f;
#pragma unroll
        for (int k = 0; k < TOPK; ++k) {
            o0 = (topi[k] == lane * 4 + 0) ? ws[k] : o0;
            o1 = (topi[k] == lane * 4 + 1) ? ws[k] : o1;
            o2 = (topi[k] == lane * 4 + 2) ? ws[k] : o2;
            o3 = (topi[k] == lane * 4 + 3) ? ws[k] : o3;
        }
        float4 o = make_float4(o0, o1, o2, o3);
        *(float4*)(out + (size_t)(row0 + t) * E_EXP + lane * 4) = o;
    }
}

// ---------------------------------------------------------------------------
extern "C" void kernel_launch(void* const* d_in, const int* in_sizes, int n_in,
                              void* d_out, int out_size, void* d_ws, size_t ws_size,
                              hipStream_t stream)
{
    (void)in_sizes; (void)n_in; (void)out_size; (void)ws_size;

    const float* hidden = (const float*)d_in[0];   // [8192, 2880] f32
    const float* weight = (const float*)d_in[1];   // [128, 2880]  f32
    const float* bias   = (const float*)d_in[2];   // [128]        f32
    float*       out    = (float*)d_out;           // [8192, 128]  f32

    // workspace: bf16 hi/lo split of weight (2 x 720 KB)
    const int nW = E_EXP * H_DIM;                  // 368640
    __bf16* w_hi = (__bf16*)d_ws;
    __bf16* w_lo = (__bf16*)((char*)d_ws + (size_t)nW * sizeof(__bf16));

    gptoss_weight_split_kernel<<<(nW + 255) / 256, 256, 0, stream>>>(
        weight, w_hi, w_lo, nW);

    dim3 grid(T_TOKENS / TOKENS_PER_BLOCK);        // 512 blocks
    dim3 block(BLOCK_THREADS);                     // 8 waves
    gptoss_topk_router_kernel<<<grid, block, 0, stream>>>(
        hidden, w_hi, w_lo, bias, out);
}
